// MLA_72679436583386
// MI455X (gfx1250) — matrix operand order repaired and checked by reading, not compiled, not run on hardware
//
#include <hip/hip_runtime.h>
#include <stdint.h>

#define NH        32
#define HEAD_DIM  128
#define SEQ       2048
#define DIM       4096
#define Q_RANK    1536
#define KV_RANK   512
#define ROPE_D    64
#define KVA_N     (KV_RANK + ROPE_D)          /* 576  */
#define KVB_N     (NH * (64 + 128))           /* 6144 */
#define RMS_EPS   1e-6f
#define ATT_SCALE 0.08838834764831845f        /* 1/sqrt(128) */

#define AS1 __attribute__((address_space(1)))
#define AS3 __attribute__((address_space(3)))

#if defined(__has_builtin)
#  if __has_builtin(__builtin_amdgcn_global_load_async_to_lds_b128) && \
      __has_builtin(__builtin_amdgcn_s_wait_asynccnt)
#    define USE_ASYNC_LDS 1
#  endif
#endif
#ifndef USE_ASYNC_LDS
#  define USE_ASYNC_LDS 0
#endif

typedef __attribute__((ext_vector_type(16))) __bf16 v16bf;
typedef __attribute__((ext_vector_type(8)))  float  v8f;
typedef __attribute__((__vector_size__(16))) int    i32x4;

union Frag16 {
    v16bf          bf;
    uint4          q[2];
    unsigned short u16[16];
};

__device__ __forceinline__ unsigned short f2bf(float f) {
    union { float f; unsigned u; } c; c.f = f;
    unsigned r = c.u + 0x7fffu + ((c.u >> 16) & 1u);   // round-to-nearest-even
    return (unsigned short)(r >> 16);
}
__device__ __forceinline__ float bf2f(unsigned short h) {
    union { unsigned u; float f; } c; c.u = ((unsigned)h) << 16;
    return c.f;
}
__device__ __forceinline__ v8f wmma_bf16(const Frag16& a, const Frag16& b, v8f c) {
    return __builtin_amdgcn_wmma_f32_16x16x32_bf16(false, a.bf, false, b.bf,
                                                   (short)0, c, false, false);
}

// ---------------------------------------------------------------------------
// fp32 -> bf16 bulk convert (4 elements / thread, 128-bit loads)
// ---------------------------------------------------------------------------
__global__ __launch_bounds__(256) void cvt_kernel(const float* __restrict__ in,
                                                  unsigned short* __restrict__ out,
                                                  long long n4) {
    long long i = (long long)blockIdx.x * blockDim.x + threadIdx.x;
    if (i >= n4) return;
    float4 v = ((const float4*)in)[i];
    unsigned lo = (unsigned)f2bf(v.x) | ((unsigned)f2bf(v.y) << 16);
    unsigned hi = (unsigned)f2bf(v.z) | ((unsigned)f2bf(v.w) << 16);
    ((uint2*)out)[i] = make_uint2(lo, hi);
}

// ---------------------------------------------------------------------------
// GEMM: C[M,N] = A[M,K](bf16) * B[N,K](bf16)^T, C fp32 or bf16.
// Block 128x128, 8 waves in 4(M) x 2(N), wave tile 32x64 (8 accumulators).
// Double-buffered LDS staging; async global->LDS DMA when available.
// ---------------------------------------------------------------------------
#define GBM 128
#define GBN 128
#define GBK 32
#define GLDT 40   /* padded row stride in bf16 elems; 80 B = 5*16 B -> b128-safe */

__global__ __launch_bounds__(256) void gemm_bf16(const unsigned short* __restrict__ A,
                                                 const unsigned short* __restrict__ B,
                                                 void* __restrict__ C,
                                                 int M, int N, int K, int out_f32) {
    __shared__ unsigned short Asm[2][GBM * GLDT];
    __shared__ unsigned short Bsm[2][GBN * GLDT];

    const int tid  = threadIdx.x;
    const int lane = tid & 31, wid = tid >> 5;
    const int wm = wid >> 1, wn = wid & 1;            // 4 x 2 waves
    const int hlf = lane >> 4, l15 = lane & 15;
    const int bm = blockIdx.y * GBM, bn = blockIdx.x * GBN;

    // staging coords: 256 thr cover 128 rows x 32 cols, 16 bf16 (2 x b128) each
    const int sr = tid >> 1, sc = (tid & 1) * 16;
    const size_t aoff = (size_t)(bm + sr) * K + sc;
    const size_t boff = (size_t)(bn + sr) * K + sc;
    const bool aok = (bm + sr) < M;
    const bool bok = (bn + sr) < N;   // OOB rows left stale; their cols never stored

    v8f acc[2][4] = {};

    auto compute = [&](int cur) {
        Frag16 af[2], bfr[4];
#pragma unroll
        for (int i = 0; i < 2; i++) {
            const unsigned short* base = Asm[cur] + (wm * 32 + i * 16 + l15) * GLDT;
            af[i].q[0] = *(const uint4*)(base + hlf * 8);
            af[i].q[1] = *(const uint4*)(base + 16 + hlf * 8);
        }
#pragma unroll
        for (int j = 0; j < 4; j++) {
            const unsigned short* base = Bsm[cur] + (wn * 64 + j * 16 + l15) * GLDT + hlf * 8;
            bfr[j].q[0] = *(const uint4*)(base);
            bfr[j].q[1] = *(const uint4*)(base + 16);
        }
#pragma unroll
        for (int i = 0; i < 2; i++)
#pragma unroll
            for (int j = 0; j < 4; j++)
                acc[i][j] = wmma_bf16(af[i], bfr[j], acc[i][j]);
    };

    const int nk = K / GBK;   // K is always a multiple of 32 here

#if USE_ASYNC_LDS
    auto issue = [&](int buf, int kb) {
        unsigned short* da = Asm[buf] + sr * GLDT + sc;
        unsigned short* db = Bsm[buf] + sr * GLDT + sc;
        if (aok) {
            __builtin_amdgcn_global_load_async_to_lds_b128(
                (AS1 i32x4*)(A + aoff + kb), (AS3 i32x4*)da, 0, 0);
            __builtin_amdgcn_global_load_async_to_lds_b128(
                (AS1 i32x4*)(A + aoff + kb), (AS3 i32x4*)da, 16, 0);
        }
        if (bok) {
            __builtin_amdgcn_global_load_async_to_lds_b128(
                (AS1 i32x4*)(B + boff + kb), (AS3 i32x4*)db, 0, 0);
            __builtin_amdgcn_global_load_async_to_lds_b128(
                (AS1 i32x4*)(B + boff + kb), (AS3 i32x4*)db, 16, 0);
        }
    };
    issue(0, 0);
    __builtin_amdgcn_s_wait_asynccnt(0);
    __syncthreads();
    for (int t = 0; t < nk; t++) {
        int cur = t & 1;
        bool more = (t + 1) < nk;
        if (more) issue(cur ^ 1, (t + 1) * GBK);   // DMA next tile, overlap with WMMA
        compute(cur);
        if (more) __builtin_amdgcn_s_wait_asynccnt(0);
        __syncthreads();
    }
#else
    uint4 ra[2], rb[2];
    auto gload = [&](int kb) {
        uint4 z = make_uint4(0, 0, 0, 0);
        ra[0] = aok ? *(const uint4*)(A + aoff + kb)     : z;
        ra[1] = aok ? *(const uint4*)(A + aoff + kb + 8) : z;
        rb[0] = bok ? *(const uint4*)(B + boff + kb)     : z;
        rb[1] = bok ? *(const uint4*)(B + boff + kb + 8) : z;
    };
    auto stst = [&](int buf) {
        *(uint4*)(Asm[buf] + sr * GLDT + sc)     = ra[0];
        *(uint4*)(Asm[buf] + sr * GLDT + sc + 8) = ra[1];
        *(uint4*)(Bsm[buf] + sr * GLDT + sc)     = rb[0];
        *(uint4*)(Bsm[buf] + sr * GLDT + sc + 8) = rb[1];
    };
    gload(0);
    stst(0);
    __syncthreads();
    for (int t = 0; t < nk; t++) {
        int cur = t & 1;
        bool more = (t + 1) < nk;
        if (more) gload((t + 1) * GBK);   // loads in flight across compute
        compute(cur);
        if (more) stst(cur ^ 1);
        __syncthreads();
    }
#endif

    // ---- epilogue: C layout lane=col(l15), vgpr r -> row r + hlf*8 ----
#pragma unroll
    for (int i = 0; i < 2; i++)
#pragma unroll
        for (int j = 0; j < 4; j++) {
            int col = bn + wn * 64 + j * 16 + l15;
            if (col >= N) continue;
#pragma unroll
            for (int r = 0; r < 8; r++) {
                int row = bm + wm * 32 + i * 16 + hlf * 8 + r;
                if (row >= M) continue;
                float v = acc[i][j][r];
                if (out_f32) ((float*)C)[(size_t)row * N + col] = v;
                else ((unsigned short*)C)[(size_t)row * N + col] = f2bf(v);
            }
        }
}

// ---------------------------------------------------------------------------
// RMSNorm over bf16 rows (fp32 stats), strided in/out
// ---------------------------------------------------------------------------
__global__ __launch_bounds__(256) void rmsnorm_kernel(const unsigned short* __restrict__ in,
                                                      const float* __restrict__ w,
                                                      unsigned short* __restrict__ out,
                                                      int n, int in_stride, int out_stride) {
    const int row = blockIdx.x;
    const unsigned short* x = in + (size_t)row * in_stride;
    unsigned short* y = out + (size_t)row * out_stride;

    float ss = 0.f;
    for (int i = threadIdx.x; i < n; i += 256) {
        float v = bf2f(x[i]);
        ss += v * v;
    }
#pragma unroll
    for (int off = 16; off > 0; off >>= 1) ss += __shfl_xor(ss, off, 32);

    __shared__ float red[8];
    if ((threadIdx.x & 31) == 0) red[threadIdx.x >> 5] = ss;
    __syncthreads();
    float tot = 0.f;
#pragma unroll
    for (int i = 0; i < 8; i++) tot += red[i];
    float scale = rsqrtf(tot / (float)n + RMS_EPS);

    for (int i = threadIdx.x; i < n; i += 256)
        y[i] = f2bf(bf2f(x[i]) * scale * w[i]);
}

// ---------------------------------------------------------------------------
// RoPE on q (in place): first 64 dims of every head
// ---------------------------------------------------------------------------
__global__ __launch_bounds__(256) void rope_q_kernel(unsigned short* __restrict__ q,
                                                     const float* __restrict__ cp,
                                                     const float* __restrict__ sp) {
    const int s = blockIdx.x;
    for (int t = threadIdx.x; t < NH * 32; t += blockDim.x) {
        int h = t >> 5, i = t & 31;
        size_t base = (size_t)s * DIM + h * HEAD_DIM + 2 * i;
        float e = bf2f(q[base]), o = bf2f(q[base + 1]);
        float c = cp[s * 32 + i], sn = sp[s * 32 + i];
        q[base]     = f2bf(e * c - o * sn);
        q[base + 1] = f2bf(e * sn + o * c);
    }
}

// ---------------------------------------------------------------------------
// Build K[s, h, 0:64]=rope(k_rope[s]) (head-broadcast), K[s,h,64:128]=k_nope
// ---------------------------------------------------------------------------
__global__ __launch_bounds__(256) void build_k_kernel(const unsigned short* __restrict__ kvr,
                                                      const unsigned short* __restrict__ kvb,
                                                      unsigned short* __restrict__ kbuf,
                                                      const float* __restrict__ cp,
                                                      const float* __restrict__ sp) {
    const int s = blockIdx.x;
    __shared__ unsigned short kr[ROPE_D];
    if (threadIdx.x < 32) {
        int i = threadIdx.x;
        float e = bf2f(kvr[(size_t)s * KVA_N + KV_RANK + 2 * i]);
        float o = bf2f(kvr[(size_t)s * KVA_N + KV_RANK + 2 * i + 1]);
        float c = cp[s * 32 + i], sn = sp[s * 32 + i];
        kr[2 * i]     = f2bf(e * c - o * sn);
        kr[2 * i + 1] = f2bf(e * sn + o * c);
    }
    __syncthreads();
    for (int t = threadIdx.x; t < NH * 64; t += blockDim.x) {
        int h = t >> 6, d = t & 63;
        kbuf[(size_t)s * DIM + h * HEAD_DIM + d]      = kr[d];
        kbuf[(size_t)s * DIM + h * HEAD_DIM + 64 + d] = kvb[(size_t)s * KVB_N + h * 64 + d];
    }
}

// ---------------------------------------------------------------------------
// Flash attention: grid (S/128, NH), 8 waves x 16 query rows, 32-key chunks.
// ---------------------------------------------------------------------------
#define KLDT 136  /* 272 B row = 17*16 B */
#define VLDT 40   /*  80 B row = 5*16 B  */
#define PLDT 40

__global__ __launch_bounds__(256) void mla_attn_kernel(const unsigned short* __restrict__ Q,
                                                       const unsigned short* __restrict__ Kb,
                                                       const unsigned short* __restrict__ KVB,
                                                       unsigned short* __restrict__ O,
                                                       float scale) {
    __shared__ unsigned short Ksm[32 * KLDT];        // [key][d]
    __shared__ unsigned short Vsm[HEAD_DIM * VLDT];  // [d][key] (transposed)
    __shared__ unsigned short Psm[8 * 16 * PLDT];    // per wave [row][k]

    const int h  = blockIdx.y;
    const int qb = blockIdx.x * 128;
    const int tid = threadIdx.x, lane = tid & 31, wid = tid >> 5;
    const int hlf = lane >> 4, l15 = lane & 15;

    // ---- resident Q fragments (4 x A 16x32 over head_dim 128) ----
    Frag16 qf[4];
    {
        const unsigned short* qp = Q + (size_t)(qb + wid * 16 + l15) * DIM + h * HEAD_DIM;
#pragma unroll
        for (int c = 0; c < 4; c++) {
            qf[c].q[0] = *(const uint4*)(qp + c * 32 + hlf * 8);
            qf[c].q[1] = *(const uint4*)(qp + c * 32 + 16 + hlf * 8);
        }
    }

    v8f o_acc[8] = {};
    float m_r[8], l_r[8];
#pragma unroll
    for (int r = 0; r < 8; r++) { m_r[r] = -1e30f; l_r[r] = 0.f; }

    const int nchunk = (qb + 128) / 32;   // causal extent
    for (int kc = 0; kc < nchunk; kc++) {
        const int k0 = kc * 32;
        __syncthreads();
        // ---- stage K chunk 32x128 ----
        {
            int r = tid >> 3, c = (tid & 7) * 16;
            const uint4* src = (const uint4*)(Kb + (size_t)(k0 + r) * DIM + h * HEAD_DIM + c);
            *(uint4*)(Ksm + r * KLDT + c)     = src[0];
            *(uint4*)(Ksm + r * KLDT + c + 8) = src[1];
        }
        // ---- stage V chunk transposed: Vsm[d][key] ----
        {
            int r = tid >> 3, c = (tid & 7) * 16;
            const unsigned short* src = KVB + (size_t)(k0 + r) * KVB_N + NH * 64 + h * HEAD_DIM + c;
            unsigned short tmp[16];
            *(uint4*)(tmp)     = ((const uint4*)src)[0];
            *(uint4*)(tmp + 8) = ((const uint4*)src)[1];
#pragma unroll
            for (int d = 0; d < 16; d++) Vsm[(c + d) * VLDT + r] = tmp[d];
        }
        __syncthreads();

        // ---- S = Q K^T : two 16x16 tiles (keys 0-15, 16-31) ----
        v8f s0 = {}, s1 = {};
#pragma unroll
        for (int c = 0; c < 4; c++) {
            Frag16 b0, b1;
            const unsigned short* k0p = Ksm + l15 * KLDT        + c * 32 + hlf * 8;
            const unsigned short* k1p = Ksm + (16 + l15) * KLDT + c * 32 + hlf * 8;
            b0.q[0] = *(const uint4*)(k0p); b0.q[1] = *(const uint4*)(k0p + 16);
            b1.q[0] = *(const uint4*)(k1p); b1.q[1] = *(const uint4*)(k1p + 16);
            s0 = wmma_bf16(qf[c], b0, s0);
            s1 = wmma_bf16(qf[c], b1, s1);
        }

        // ---- online softmax (rows live as vgpr r + half*8 across 16 lanes) ----
        const int rowg = qb + wid * 16;
#pragma unroll
        for (int r = 0; r < 8; r++) {
            int row = rowg + hlf * 8 + r;
            float v0 = s0[r] * scale, v1 = s1[r] * scale;
            if (k0 + l15 > row)      v0 = -1e30f;
            if (k0 + 16 + l15 > row) v1 = -1e30f;
            float mx = fmaxf(v0, v1);
#pragma unroll
            for (int off = 1; off < 16; off <<= 1) mx = fmaxf(mx, __shfl_xor(mx, off, 32));
            float m_new = fmaxf(m_r[r], mx);
            float alpha = __expf(m_r[r] - m_new);
            float p0 = __expf(v0 - m_new), p1 = __expf(v1 - m_new);
            float ps = p0 + p1;
#pragma unroll
            for (int off = 1; off < 16; off <<= 1) ps += __shfl_xor(ps, off, 32);
            l_r[r] = l_r[r] * alpha + ps;
            m_r[r] = m_new;
#pragma unroll
            for (int j = 0; j < 8; j++) o_acc[j][r] *= alpha;
            unsigned short* prow = Psm + (wid * 16 + hlf * 8 + r) * PLDT;
            prow[l15]      = f2bf(p0);
            prow[16 + l15] = f2bf(p1);
        }
        __syncthreads();

        // ---- PV: P(16x32) as A-frag, V^T rows as B-frags ----
        Frag16 pa;
        {
            const unsigned short* pb = Psm + (wid * 16 + l15) * PLDT;
            pa.q[0] = *(const uint4*)(pb + hlf * 8);
            pa.q[1] = *(const uint4*)(pb + 16 + hlf * 8);
        }
#pragma unroll
        for (int j = 0; j < 8; j++) {
            Frag16 vb;
            const unsigned short* vp = Vsm + (j * 16 + l15) * VLDT + hlf * 8;
            vb.q[0] = *(const uint4*)(vp);
            vb.q[1] = *(const uint4*)(vp + 16);
            o_acc[j] = wmma_bf16(pa, vb, o_acc[j]);
        }
    }

    // ---- epilogue: normalize and store ----
#pragma unroll
    for (int r = 0; r < 8; r++) {
        int row = qb + wid * 16 + hlf * 8 + r;
        float inv = 1.0f / l_r[r];
#pragma unroll
        for (int j = 0; j < 8; j++)
            O[(size_t)row * DIM + h * HEAD_DIM + j * 16 + l15] = f2bf(o_acc[j][r] * inv);
    }
}

// ---------------------------------------------------------------------------
// Launch
// ---------------------------------------------------------------------------
extern "C" void kernel_launch(void* const* d_in, const int* in_sizes, int n_in,
                              void* d_out, int out_size, void* d_ws, size_t ws_size,
                              hipStream_t stream) {
    (void)in_sizes; (void)n_in; (void)out_size; (void)ws_size;

    const float* x     = (const float*)d_in[0];
    const float* wq_a  = (const float*)d_in[1];
    const float* q_ln  = (const float*)d_in[2];
    const float* wq_b  = (const float*)d_in[3];
    const float* wkv_a = (const float*)d_in[4];
    const float* kv_ln = (const float*)d_in[5];
    const float* wkv_b = (const float*)d_in[6];
    const float* wo    = (const float*)d_in[7];
    const float* fcos  = (const float*)d_in[8];
    const float* fsin  = (const float*)d_in[9];
    float* out = (float*)d_out;

    char* ws = (char*)d_ws;
    size_t off = 0;
    auto alloc = [&](size_t elems) -> unsigned short* {
        unsigned short* p = (unsigned short*)(ws + off);
        off = (off + elems * 2 + 255) & ~(size_t)255;
        return p;
    };
    unsigned short* xb    = alloc((size_t)SEQ * DIM);
    unsigned short* wqab  = alloc((size_t)Q_RANK * DIM);
    unsigned short* wqbb  = alloc((size_t)DIM * Q_RANK);
    unsigned short* wkvab = alloc((size_t)KVA_N * DIM);
    unsigned short* wkvbb = alloc((size_t)KVB_N * KV_RANK);
    unsigned short* wob   = alloc((size_t)DIM * DIM);
    unsigned short* qa    = alloc((size_t)SEQ * Q_RANK);
    unsigned short* qbuf  = alloc((size_t)SEQ * DIM);
    unsigned short* kvr   = alloc((size_t)SEQ * KVA_N);
    unsigned short* kvan  = alloc((size_t)SEQ * KV_RANK);
    unsigned short* kvb   = alloc((size_t)SEQ * KVB_N);
    unsigned short* kbuf  = alloc((size_t)SEQ * DIM);
    unsigned short* attn  = alloc((size_t)SEQ * DIM);

    auto cvt = [&](const float* src, unsigned short* dst, size_t n) {
        long long n4 = (long long)(n / 4);
        cvt_kernel<<<dim3((unsigned)((n4 + 255) / 256)), dim3(256), 0, stream>>>(src, dst, n4);
    };
    cvt(x,     xb,    (size_t)SEQ * DIM);
    cvt(wq_a,  wqab,  (size_t)Q_RANK * DIM);
    cvt(wq_b,  wqbb,  (size_t)DIM * Q_RANK);
    cvt(wkv_a, wkvab, (size_t)KVA_N * DIM);
    cvt(wkv_b, wkvbb, (size_t)KVB_N * KV_RANK);
    cvt(wo,    wob,   (size_t)DIM * DIM);

    // q_a = x @ wq_a^T ; rmsnorm
    gemm_bf16<<<dim3(Q_RANK / GBN, SEQ / GBM), 256, 0, stream>>>(xb, wqab, qa, SEQ, Q_RANK, DIM, 0);
    rmsnorm_kernel<<<SEQ, 256, 0, stream>>>(qa, q_ln, qa, Q_RANK, Q_RANK, Q_RANK);
    // q = q_a @ wq_b^T
    gemm_bf16<<<dim3(DIM / GBN, SEQ / GBM), 256, 0, stream>>>(qa, wqbb, qbuf, SEQ, DIM, Q_RANK, 0);
    // kv = x @ wkv_a^T (576 wide) ; rmsnorm first 512
    gemm_bf16<<<dim3((KVA_N + GBN - 1) / GBN, SEQ / GBM), 256, 0, stream>>>(xb, wkvab, kvr, SEQ, KVA_N, DIM, 0);
    rmsnorm_kernel<<<SEQ, 256, 0, stream>>>(kvr, kv_ln, kvan, KV_RANK, KVA_N, KV_RANK);
    // kv_b = kv_a @ wkv_b^T (k_nope | v)
    gemm_bf16<<<dim3(KVB_N / GBN, SEQ / GBM), 256, 0, stream>>>(kvan, wkvbb, kvb, SEQ, KVB_N, KV_RANK, 0);
    // rope + K assembly
    rope_q_kernel<<<SEQ, 256, 0, stream>>>(qbuf, fcos, fsin);
    build_k_kernel<<<SEQ, 256, 0, stream>>>(kvr, kvb, kbuf, fcos, fsin);
    // attention
    mla_attn_kernel<<<dim3(SEQ / 128, NH), 256, 0, stream>>>(qbuf, kbuf, kvb, attn, ATT_SCALE);
    // out = attn @ wo^T (fp32 to d_out)
    gemm_bf16<<<dim3(DIM / GBN, SEQ / GBM), 256, 0, stream>>>(attn, wob, out, SEQ, DIM, DIM, 1);
}
